// SDMamba_73358041416056
// MI455X (gfx1250) — matrix operand order repaired and checked by reading, not compiled, not run on hardware
//
#include <hip/hip_runtime.h>
#include <hip/hip_bf16.h>

typedef __attribute__((ext_vector_type(16))) __bf16 v16bf;
typedef __attribute__((ext_vector_type(8)))  float  v8f;
typedef __attribute__((ext_vector_type(4)))  unsigned int uint4v;

#define B_SZ     32
#define L_HISTC  96
#define FUTC     96
#define N_SER    862
#define D_MODELC 512
#define D_STATEC 16
#define D_INNERC 1024
#define DT_RANKC 32
#define D_FFC    512
#define TOK      (B_SZ * N_SER)      // 27584 tokens
#define EPSF     1e-5f

#define GEMM_RELU 1
#define GEMM_ACC  2

// ----------------------------------------------------------------------------
// f32 -> bf16 conversion
// ----------------------------------------------------------------------------
__global__ void cvt_f32_bf16(const float* __restrict__ in, __bf16* __restrict__ out, int n) {
  int i = blockIdx.x * blockDim.x + threadIdx.x;
  if (i < n) out[i] = (__bf16)in[i];
}

// ----------------------------------------------------------------------------
// Async global->LDS staging helpers (CDNA5 VGPR-bypassing copy, ASYNCcnt)
// lds_addr: byte address in LDS space (low 32 bits of generic address of a
// __shared__ object are the LDS offset per the aperture mapping).
// ----------------------------------------------------------------------------
__device__ __forceinline__ void async_copy_b128(unsigned lds_addr, const void* gptr) {
  asm volatile("global_load_async_to_lds_b128 %0, %1, off"
               :: "v"(lds_addr), "v"((unsigned long long)(uintptr_t)gptr)
               : "memory");
}
__device__ __forceinline__ void wait_async0() {
  asm volatile("s_wait_asynccnt 0x0" ::: "memory");
}

// ----------------------------------------------------------------------------
// WMMA GEMM: C[M,Nn] = A[M,K](bf16,row-major,lda) x Bw[Nn,K](bf16,row-major,ldb)^T
// epilogue: +bias[n], relu, accumulate-into-C, optional bf16 shadow store.
// Block: 256 threads = 8 waves; block tile 128x128; wave tile 32x64 (2x4 WMMA).
// Full tiles are staged with global_load_async_to_lds_b128; tail tiles use the
// guarded load + ds_store path with zero fill.
// ----------------------------------------------------------------------------
__global__ __launch_bounds__(256)
void gemm_bf16_wmma(const __bf16* __restrict__ A, int lda,
                    const __bf16* __restrict__ Bw, int ldb,
                    float* __restrict__ C, int ldc,
                    const float* __restrict__ bias,
                    __bf16* __restrict__ Cbf,
                    int M, int Nn, int K, int flags) {
  __shared__ __align__(16) __bf16 As[128][40];
  __shared__ __align__(16) __bf16 Bs[128][40];

  const int tid  = threadIdx.x;
  const int wave = tid >> 5;
  const int lane = tid & 31;
  const int l15  = lane & 15;
  const int lh   = lane >> 4;          // lane half: 0 for 0-15, 1 for 16-31
  const int bm   = blockIdx.x * 128;
  const int bn   = blockIdx.y * 128;
  const int wm   = (wave >> 1) * 32;   // 4 waves along M
  const int wn   = (wave & 1) * 64;    // 2 waves along N, 64 wide each

  const bool fullA = (bm + 128 <= M);
  const bool fullB = (bn + 128 <= Nn);

  v8f zero = {0.f,0.f,0.f,0.f,0.f,0.f,0.f,0.f};
  v8f cacc[2][4];
  #pragma unroll
  for (int mi = 0; mi < 2; ++mi)
    #pragma unroll
    for (int ni = 0; ni < 4; ++ni) cacc[mi][ni] = zero;

  union Frag { v16bf v; uint4v u[2]; };
  const uint4v uz = {0u,0u,0u,0u};

  const unsigned asBase = (unsigned)(uintptr_t)&As[0][0];
  const unsigned bsBase = (unsigned)(uintptr_t)&Bs[0][0];

  for (int kk = 0; kk < K; kk += 32) {
    // ---- stage A tile (128x32 bf16): 512 x 16B segments, 2 per thread ----
    if (fullA) {
      #pragma unroll
      for (int i = 0; i < 2; ++i) {
        int idx = tid + i * 256;
        int r   = idx >> 2;
        int c8  = (idx & 3) * 8;
        async_copy_b128(asBase + (unsigned)(r * 40 + c8) * 2,
                        &A[(size_t)(bm + r) * lda + kk + c8]);
      }
    } else {
      #pragma unroll
      for (int i = 0; i < 2; ++i) {
        int idx = tid + i * 256;
        int r   = idx >> 2;
        int c8  = (idx & 3) * 8;
        int gm  = bm + r;
        uint4v val = uz;
        if (gm < M) {
          val = *(const uint4v*)&A[(size_t)gm * lda + kk + c8];
          if (kk + 32 < K) __builtin_prefetch(&A[(size_t)gm * lda + kk + 32 + c8], 0, 3);
        }
        *(uint4v*)&As[r][c8] = val;
      }
    }
    // ---- stage B tile (128x32 bf16): 512 x 16B segments, 2 per thread ----
    if (fullB) {
      #pragma unroll
      for (int i = 0; i < 2; ++i) {
        int idx = tid + i * 256;
        int r   = idx >> 2;
        int c8  = (idx & 3) * 8;
        async_copy_b128(bsBase + (unsigned)(r * 40 + c8) * 2,
                        &Bw[(size_t)(bn + r) * ldb + kk + c8]);
      }
    } else {
      #pragma unroll
      for (int i = 0; i < 2; ++i) {
        int idx = tid + i * 256;
        int r   = idx >> 2;
        int c8  = (idx & 3) * 8;
        int gn  = bn + r;
        uint4v val = uz;
        if (gn < Nn) {
          val = *(const uint4v*)&Bw[(size_t)gn * ldb + kk + c8];
          if (kk + 32 < K) __builtin_prefetch(&Bw[(size_t)gn * ldb + kk + 32 + c8], 0, 3);
        }
        *(uint4v*)&Bs[r][c8] = val;
      }
    }
    wait_async0();
    __syncthreads();

    // ---- build fragments (contiguous k order k = 16*lh + s, same for A and B) ----
    // A 16x32: lanes 0-15 hold K 0-15, lanes 16-31 hold K 16-31 (M = lane%16)
    Frag fa0, fa1, fb0, fb1, fb2, fb3;
    fa0.u[0] = *(const uint4v*)&As[wm +      l15][lh * 16];
    fa0.u[1] = *(const uint4v*)&As[wm +      l15][lh * 16 + 8];
    fa1.u[0] = *(const uint4v*)&As[wm + 16 + l15][lh * 16];
    fa1.u[1] = *(const uint4v*)&As[wm + 16 + l15][lh * 16 + 8];
    // B 32x16: lanes 0-15 hold K 0-15, lanes 16-31 hold K 16-31 (N = lane%16)
    fb0.u[0] = *(const uint4v*)&Bs[wn +      l15][lh * 16];
    fb0.u[1] = *(const uint4v*)&Bs[wn +      l15][lh * 16 + 8];
    fb1.u[0] = *(const uint4v*)&Bs[wn + 16 + l15][lh * 16];
    fb1.u[1] = *(const uint4v*)&Bs[wn + 16 + l15][lh * 16 + 8];
    fb2.u[0] = *(const uint4v*)&Bs[wn + 32 + l15][lh * 16];
    fb2.u[1] = *(const uint4v*)&Bs[wn + 32 + l15][lh * 16 + 8];
    fb3.u[0] = *(const uint4v*)&Bs[wn + 48 + l15][lh * 16];
    fb3.u[1] = *(const uint4v*)&Bs[wn + 48 + l15][lh * 16 + 8];

    cacc[0][0] = __builtin_amdgcn_wmma_f32_16x16x32_bf16(false, fa0.v, false, fb0.v,
                 (short)0, cacc[0][0], false, false);
    cacc[0][1] = __builtin_amdgcn_wmma_f32_16x16x32_bf16(false, fa0.v, false, fb1.v,
                 (short)0, cacc[0][1], false, false);
    cacc[0][2] = __builtin_amdgcn_wmma_f32_16x16x32_bf16(false, fa0.v, false, fb2.v,
                 (short)0, cacc[0][2], false, false);
    cacc[0][3] = __builtin_amdgcn_wmma_f32_16x16x32_bf16(false, fa0.v, false, fb3.v,
                 (short)0, cacc[0][3], false, false);
    cacc[1][0] = __builtin_amdgcn_wmma_f32_16x16x32_bf16(false, fa1.v, false, fb0.v,
                 (short)0, cacc[1][0], false, false);
    cacc[1][1] = __builtin_amdgcn_wmma_f32_16x16x32_bf16(false, fa1.v, false, fb1.v,
                 (short)0, cacc[1][1], false, false);
    cacc[1][2] = __builtin_amdgcn_wmma_f32_16x16x32_bf16(false, fa1.v, false, fb2.v,
                 (short)0, cacc[1][2], false, false);
    cacc[1][3] = __builtin_amdgcn_wmma_f32_16x16x32_bf16(false, fa1.v, false, fb3.v,
                 (short)0, cacc[1][3], false, false);
    __syncthreads();
  }

  // ---- epilogue: C 16x16 layout: VGPR r -> row (lh*8 + r), col = lane%16 ----
  const bool relu  = (flags & GEMM_RELU) != 0;
  const bool accum = (flags & GEMM_ACC)  != 0;
  #pragma unroll
  for (int mi = 0; mi < 2; ++mi) {
    #pragma unroll
    for (int ni = 0; ni < 4; ++ni) {
      int gn = bn + wn + ni * 16 + l15;
      #pragma unroll
      for (int r = 0; r < 8; ++r) {
        int gm = bm + wm + mi * 16 + lh * 8 + r;
        if (gm < M && gn < Nn) {
          float v = cacc[mi][ni][r];
          if (bias)  v += bias[gn];
          if (relu)  v = fmaxf(v, 0.f);
          size_t off = (size_t)gm * ldc + gn;
          if (accum) v += C[off];
          C[off] = v;
          if (Cbf) Cbf[off] = (__bf16)v;
        }
      }
    }
  }
}

// ----------------------------------------------------------------------------
// RevIN stats + normalized history in (b,n,l) bf16 layout for embed GEMM
// ----------------------------------------------------------------------------
__global__ void revin_prep(const float* __restrict__ hist,   // (B, L, N)
                           const float* __restrict__ rw, const float* __restrict__ rb,
                           float* __restrict__ meanp, float* __restrict__ stdp,
                           __bf16* __restrict__ xn) {        // (B*N, L)
  int idx = blockIdx.x * blockDim.x + threadIdx.x;
  if (idx >= TOK) return;
  int b = idx / N_SER, n = idx - b * N_SER;
  const float* p = hist + (size_t)b * L_HISTC * N_SER + n;
  float s = 0.f, s2 = 0.f;
  for (int l = 0; l < L_HISTC; ++l) { float v = p[(size_t)l * N_SER]; s += v; s2 += v * v; }
  float m   = s / L_HISTC;
  float var = s2 / L_HISTC - m * m;
  float sd  = sqrtf(var + EPSF);
  meanp[idx] = m; stdp[idx] = sd;
  float w = rw[n], bb = rb[n];
  __bf16* q = xn + (size_t)idx * L_HISTC;
  for (int l = 0; l < L_HISTC; ++l) {
    float v = p[(size_t)l * N_SER];
    q[l] = (__bf16)(((v - m) / sd) * w + bb);
  }
}

// ----------------------------------------------------------------------------
// Depthwise causal conv (D_CONV=4) over the series axis + SiLU. dir=1 reversed.
// ----------------------------------------------------------------------------
__global__ void conv_silu(const float* __restrict__ xz,   // (T, 2048), xi = cols 0..1023
                          const float* __restrict__ w,    // (1024, 4)
                          const float* __restrict__ bias, // (1024)
                          float* __restrict__ xc, __bf16* __restrict__ xcb,
                          int dir) {
  int idx = blockIdx.x * blockDim.x + threadIdx.x;
  if (idx >= TOK * D_INNERC) return;
  int d = idx & (D_INNERC - 1);
  int t = idx >> 10;
  int b = t / N_SER, n = t - b * N_SER;
  float acc = bias[d];
  #pragma unroll
  for (int j = 0; j < 4; ++j) {
    int src = (dir == 0) ? (n - 3 + j) : (n + 3 - j);
    if (src >= 0 && src < N_SER)
      acc += w[d * 4 + j] * xz[((size_t)(b * N_SER + src)) * 2048 + d];
  }
  float y = acc / (1.f + __expf(-acc));   // silu
  xc[idx]  = y;
  xcb[idx] = (__bf16)y;
}

// ----------------------------------------------------------------------------
// Selective scan: one thread per (b, d); h[16] in registers; dir flag.
// Fuses softplus(dt)+bias, SSM recurrence, D-skip, and silu(z) gate.
// ----------------------------------------------------------------------------
__global__ __launch_bounds__(256)
void mamba_scan(const float* __restrict__ dtraw,  // (T,1024)  dt @ dt_proj^T (no bias)
                const float* __restrict__ dbc,    // (T,64)    [dt_lowrank | B | C]
                const float* __restrict__ xc,     // (T,1024)  conv+silu output
                const float* __restrict__ xz,     // (T,2048)  z = cols 1024..2047
                const float* __restrict__ A_log,  // (1024,16)
                const float* __restrict__ Dp,     // (1024)
                const float* __restrict__ dtb,    // (1024)
                __bf16* __restrict__ y,           // (T,1024)  gated output (bf16)
                int dir) {
  int idx = blockIdx.x * blockDim.x + threadIdx.x;
  if (idx >= B_SZ * D_INNERC) return;
  int d = idx & (D_INNERC - 1);
  int b = idx >> 10;
  float Am[D_STATEC], h[D_STATEC];
  #pragma unroll
  for (int s = 0; s < D_STATEC; ++s) { Am[s] = -__expf(A_log[d * D_STATEC + s]); h[s] = 0.f; }
  const float Dd = Dp[d], bia = dtb[d];
  for (int step = 0; step < N_SER; ++step) {
    int n = (dir == 0) ? step : (N_SER - 1 - step);
    size_t t = (size_t)b * N_SER + n;
    float dtv = dtraw[t * D_INNERC + d] + bia;
    dtv = (dtv > 20.f) ? dtv : log1pf(__expf(dtv));       // softplus
    float xt = xc[t * D_INNERC + d];
    float dx = dtv * xt;
    const float* Bc = dbc + t * 64;
    float yv = 0.f;
    #pragma unroll
    for (int s = 0; s < D_STATEC; ++s) {
      h[s] = __expf(dtv * Am[s]) * h[s] + dx * Bc[32 + s];
      yv  += h[s] * Bc[48 + s];
    }
    yv += xt * Dd;
    float z = xz[t * 2048 + D_INNERC + d];
    yv *= z / (1.f + __expf(-z));                         // * silu(z)
    y[t * D_INNERC + d] = (__bf16)yv;
  }
}

// ----------------------------------------------------------------------------
// LayerNorm over D_MODEL=512: one wave per row, wave-reduce via shfl (wave32)
// ----------------------------------------------------------------------------
__global__ __launch_bounds__(256)
void layer_norm(const float* __restrict__ x, const float* __restrict__ w,
                const float* __restrict__ bb,
                float* __restrict__ out, __bf16* __restrict__ outb, int rows) {
  int wid  = (blockIdx.x * blockDim.x + threadIdx.x) >> 5;
  int lane = threadIdx.x & 31;
  if (wid >= rows) return;
  const float* p = x + (size_t)wid * D_MODELC;
  float s = 0.f, s2 = 0.f;
  for (int i = lane; i < D_MODELC; i += 32) { float v = p[i]; s += v; s2 += v * v; }
  #pragma unroll
  for (int off = 16; off > 0; off >>= 1) {
    s  += __shfl_xor(s,  off, 32);
    s2 += __shfl_xor(s2, off, 32);
  }
  float m   = s / D_MODELC;
  float var = s2 / D_MODELC - m * m;
  float inv = rsqrtf(var + EPSF);
  for (int i = lane; i < D_MODELC; i += 32) {
    float v = (p[i] - m) * inv * w[i] + bb[i];
    out[(size_t)wid * D_MODELC + i] = v;
    if (outb) outb[(size_t)wid * D_MODELC + i] = (__bf16)v;
  }
}

// ----------------------------------------------------------------------------
// De-normalize + transpose (b,n,f) -> (b,f,n)
// ----------------------------------------------------------------------------
__global__ void denorm_out(const float* __restrict__ dec,   // (T, FUT)
                           const float* __restrict__ rw, const float* __restrict__ rb,
                           const float* __restrict__ meanp, const float* __restrict__ stdp,
                           float* __restrict__ out) {        // (B, FUT, N)
  int idx = blockIdx.x * blockDim.x + threadIdx.x;
  if (idx >= B_SZ * FUTC * N_SER) return;
  int n = idx % N_SER;
  int f = (idx / N_SER) % FUTC;
  int b = idx / (FUTC * N_SER);
  float v = dec[((size_t)b * N_SER + n) * FUTC + f];
  v = (v - rb[n]) / (rw[n] + EPSF * EPSF) * stdp[b * N_SER + n] + meanp[b * N_SER + n];
  out[idx] = v;
}

// ----------------------------------------------------------------------------
// Host side
// ----------------------------------------------------------------------------
static void launch_gemm(hipStream_t s, const __bf16* A, int lda, const __bf16* Bw, int ldb,
                        float* C, int ldc, const float* bias, __bf16* Cbf,
                        int M, int Nn, int K, int flags) {
  dim3 g((unsigned)((M + 127) / 128), (unsigned)((Nn + 127) / 128));
  gemm_bf16_wmma<<<g, 256, 0, s>>>(A, lda, Bw, ldb, C, ldc, bias, Cbf, M, Nn, K, flags);
}

static void launch_cvt(hipStream_t s, const float* in, __bf16* out, int n) {
  cvt_f32_bf16<<<(n + 255) / 256, 256, 0, s>>>(in, out, n);
}

extern "C" void kernel_launch(void* const* d_in, const int* in_sizes, int n_in,
                              void* d_out, int out_size, void* d_ws, size_t ws_size,
                              hipStream_t stream) {
  (void)in_sizes; (void)n_in; (void)out_size; (void)ws_size;
  const float* hist      = (const float*)d_in[0];
  const float* revin_w   = (const float*)d_in[1];
  const float* revin_b   = (const float*)d_in[2];
  const float* embed_w   = (const float*)d_in[3];
  const float* embed_b   = (const float*)d_in[4];
  const float* in_proj_w = (const float*)d_in[5];
  const float* conv_w    = (const float*)d_in[6];
  const float* conv_b    = (const float*)d_in[7];
  const float* x_proj_w  = (const float*)d_in[8];
  const float* dt_proj_w = (const float*)d_in[9];
  const float* dt_proj_b = (const float*)d_in[10];
  const float* A_log     = (const float*)d_in[11];
  const float* Dp        = (const float*)d_in[12];
  const float* out_proj_w= (const float*)d_in[13];
  const float* conv1_w   = (const float*)d_in[14];
  const float* conv1_b   = (const float*)d_in[15];
  const float* conv2_w   = (const float*)d_in[16];
  const float* conv2_b   = (const float*)d_in[17];
  const float* norm1_w   = (const float*)d_in[18];
  const float* norm1_b   = (const float*)d_in[19];
  const float* norm2_w   = (const float*)d_in[20];
  const float* norm2_b   = (const float*)d_in[21];
  const float* normf_w   = (const float*)d_in[22];
  const float* normf_b   = (const float*)d_in[23];
  const float* proj_w    = (const float*)d_in[24];
  const float* proj_b    = (const float*)d_in[25];
  float* out = (float*)d_out;

  // ---- bump allocator over workspace ----
  char* ws = (char*)d_ws;
  size_t off = 0;
  auto alloc = [&](size_t bytes) -> void* {
    void* p = ws + off;
    off += (bytes + 255) & ~(size_t)255;
    return p;
  };
  const size_t T = TOK;
  float*  meanp  = (float*)alloc(T * 4);
  float*  stdp   = (float*)alloc(T * 4);
  __bf16* xn     = (__bf16*)alloc(T * L_HISTC * 2);
  float*  e_f    = (float*)alloc(T * D_MODELC * 4);
  __bf16* e_b    = (__bf16*)alloc(T * D_MODELC * 2);
  float*  acc    = (float*)alloc(T * D_MODELC * 4);
  float*  xz     = (float*)alloc(T * 2048 * 4);
  float*  xc     = (float*)alloc(T * D_INNERC * 4);
  __bf16* xcb    = (__bf16*)alloc(T * D_INNERC * 2);
  float*  dbc    = (float*)alloc(T * 64 * 4);
  __bf16* dbcb   = (__bf16*)alloc(T * 64 * 2);
  float*  dtfull = (float*)alloc(T * D_INNERC * 4);
  __bf16* yb     = (__bf16*)alloc(T * D_INNERC * 2);
  float*  ffn1   = (float*)alloc(T * D_FFC * 4);
  __bf16* ffn1b  = (__bf16*)alloc(T * D_FFC * 2);
  float*  dec    = (float*)alloc(T * FUTC * 4);
  // bf16 weight shadows
  __bf16* embedWb = (__bf16*)alloc(512 * 96 * 2);
  __bf16* inWb    = (__bf16*)alloc((size_t)4 * 2048 * 512 * 2);
  __bf16* xpWb    = (__bf16*)alloc((size_t)4 * 64 * 1024 * 2);
  __bf16* dtWb    = (__bf16*)alloc((size_t)4 * 1024 * 32 * 2);
  __bf16* outWb   = (__bf16*)alloc((size_t)4 * 512 * 1024 * 2);
  __bf16* c1Wb    = (__bf16*)alloc((size_t)2 * 512 * 512 * 2);
  __bf16* c2Wb    = (__bf16*)alloc((size_t)2 * 512 * 512 * 2);
  __bf16* projWb  = (__bf16*)alloc(96 * 512 * 2);

  // ---- weight conversion (deterministic every call) ----
  launch_cvt(stream, embed_w,    embedWb, 512 * 96);
  launch_cvt(stream, in_proj_w,  inWb,    4 * 2048 * 512);
  launch_cvt(stream, x_proj_w,   xpWb,    4 * 64 * 1024);
  launch_cvt(stream, dt_proj_w,  dtWb,    4 * 1024 * 32);
  launch_cvt(stream, out_proj_w, outWb,   4 * 512 * 1024);
  launch_cvt(stream, conv1_w,    c1Wb,    2 * 512 * 512);
  launch_cvt(stream, conv2_w,    c2Wb,    2 * 512 * 512);
  launch_cvt(stream, proj_w,     projWb,  96 * 512);

  // ---- RevIN + embed ----
  revin_prep<<<(TOK + 255) / 256, 256, 0, stream>>>(hist, revin_w, revin_b, meanp, stdp, xn);
  launch_gemm(stream, xn, L_HISTC, embedWb, L_HISTC, e_f, D_MODELC, embed_b, e_b,
              TOK, D_MODELC, L_HISTC, 0);

  const int nLNwaves = (TOK * 32 + 255) / 256;
  for (int l = 0; l < 2; ++l) {
    hipMemcpyAsync(acc, e_f, (size_t)T * D_MODELC * 4, hipMemcpyDeviceToDevice, stream);
    for (int r = 0; r < 2; ++r) {
      const int lr = l * 2 + r;
      // in_proj: xz = e @ W^T   (T x 2048)
      launch_gemm(stream, e_b, D_MODELC, inWb + (size_t)lr * 2048 * 512, D_MODELC,
                  xz, 2048, nullptr, nullptr, TOK, 2048, D_MODELC, 0);
      // depthwise conv + silu (direction-aware)
      conv_silu<<<(TOK * D_INNERC + 255) / 256, 256, 0, stream>>>(
          xz, conv_w + (size_t)lr * D_INNERC * 4, conv_b + (size_t)lr * D_INNERC,
          xc, xcb, r);
      // x_proj: dbc = xc @ W^T   (T x 64)
      launch_gemm(stream, xcb, D_INNERC, xpWb + (size_t)lr * 64 * 1024, D_INNERC,
                  dbc, 64, nullptr, dbcb, TOK, 64, D_INNERC, 0);
      // dt_proj: dtfull = dbc[:, :32] @ W^T   (T x 1024); lda=64, K=32
      launch_gemm(stream, dbcb, 64, dtWb + (size_t)lr * 1024 * 32, 32,
                  dtfull, D_INNERC, nullptr, nullptr, TOK, D_INNERC, DT_RANKC, 0);
      // selective scan + D-skip + silu(z) gate
      mamba_scan<<<(B_SZ * D_INNERC + 255) / 256, 256, 0, stream>>>(
          dtfull, dbc, xc, xz,
          A_log + (size_t)lr * D_INNERC * D_STATEC,
          Dp + (size_t)lr * D_INNERC,
          dt_proj_b + (size_t)lr * D_INNERC, yb, r);
      // out_proj, accumulated into residual
      launch_gemm(stream, yb, D_INNERC, outWb + (size_t)lr * 512 * 1024, D_INNERC,
                  acc, D_MODELC, nullptr, nullptr, TOK, D_MODELC, D_INNERC, GEMM_ACC);
    }
    layer_norm<<<nLNwaves, 256, 0, stream>>>(acc, norm1_w + l * 512, norm1_b + l * 512,
                                             e_f, e_b, TOK);
    // FFN
    hipMemcpyAsync(acc, e_f, (size_t)T * D_MODELC * 4, hipMemcpyDeviceToDevice, stream);
    launch_gemm(stream, e_b, D_MODELC, c1Wb + (size_t)l * 512 * 512, D_MODELC,
                ffn1, D_FFC, conv1_b + l * 512, ffn1b, TOK, D_FFC, D_MODELC, GEMM_RELU);
    launch_gemm(stream, ffn1b, D_FFC, c2Wb + (size_t)l * 512 * 512, D_FFC,
                acc, D_MODELC, conv2_b + l * 512, nullptr, TOK, D_MODELC, D_FFC, GEMM_ACC);
    layer_norm<<<nLNwaves, 256, 0, stream>>>(acc, norm2_w + l * 512, norm2_b + l * 512,
                                             e_f, e_b, TOK);
  }
  // final LN + projection head
  layer_norm<<<nLNwaves, 256, 0, stream>>>(e_f, normf_w, normf_b, e_f, e_b, TOK);
  launch_gemm(stream, e_b, D_MODELC, projWb, D_MODELC, dec, FUTC, proj_b, nullptr,
              TOK, FUTC, D_MODELC, 0);
  denorm_out<<<(B_SZ * FUTC * N_SER + 255) / 256, 256, 0, stream>>>(
      dec, revin_w, revin_b, meanp, stdp, out);
}
